// VectorQuantizerGumbelSoftmax_14654428413995
// MI455X (gfx1250) — matrix operand order repaired and checked by reading, not compiled, not run on hardware
//
#include <hip/hip_runtime.h>
#include <hip/hip_bf16.h>
#include <cstdint>
#include <cstddef>

typedef _Float16 v16h __attribute__((ext_vector_type(16)));
typedef _Float16 v8h  __attribute__((ext_vector_type(8)));
typedef float    v8f  __attribute__((ext_vector_type(8)));
typedef unsigned int u32x4 __attribute__((ext_vector_type(4)));
typedef int      i32x8 __attribute__((ext_vector_type(8)));
typedef int      i32x4 __attribute__((ext_vector_type(4)));

namespace {
constexpr int  kNE   = 1024;
constexpr int  kDim  = 64;
constexpr int  kRows = 131072;                 // 32 * 64 * 64
constexpr int  kRowsPerWave = 16;
constexpr int  kWavesPerWG  = 8;
constexpr int  kRowsPerWG   = kRowsPerWave * kWavesPerWG;   // 128
constexpr int  kNumWG = kRows / kRowsPerWG;                 // 1024

// ---- output layout (floats, concatenated tuple) ----
constexpr size_t OFF_LOSS = 0;
constexpr size_t OFF_ZQ   = 1;
constexpr size_t OFF_PERP = OFF_ZQ + (size_t)kRows * kDim;   // 8388609
constexpr size_t OFF_OH   = OFF_PERP + 1;                    // 8388610
constexpr size_t OFF_IDX  = OFF_OH + (size_t)kRows * kNE;    // 142606338

// ---- workspace layout (bytes) ----
constexpr size_t WS_EMBH  = 0;        // 1024*64 f16  (codebook, code-major)
constexpr size_t WS_EMBT  = 131072;   // 64*1024 f16  (codebook, dim-major)
constexpr size_t WS_EMBSQ = 262144;   // 1024 f32     (||e||^2)
constexpr size_t WS_AVGP  = 266240;   // 1024 f32     (column-sum accumulator)
constexpr size_t WS_LOSS  = 270336;   // 1 f32        (sum of squared diffs)

// ---- dynamic LDS layout (bytes) ----
constexpr unsigned LDS_PT   = 0;        // 8 waves * 16*1024 f16 = 262144
constexpr unsigned LDS_ZT   = 262144;   // 8 waves * 16*64  f32 = 32768
constexpr unsigned LDS_AVGP = 294912;   // 1024 f32 = 4096
constexpr unsigned LDS_RINV = 299008;   // 8 waves * 16 f32 = 512
constexpr unsigned LDS_TOTAL = 299520;  // < 320KB per WGP
} // namespace

__device__ __forceinline__ v8f wmma_f16(v16h a, v16h b, v8f c) {
  // D = A(16x32 f16) x B(32x16 f16) + C(16x16 f32)
  return __builtin_amdgcn_wmma_f32_16x16x32_f16(
      /*neg_a=*/false, a, /*neg_b=*/false, b,
      /*c_mod=*/(short)0, c, /*reuse_a=*/false, /*reuse_b=*/false);
}

// ---------------------------------------------------------------------------
// Prep: f16 codebook in both majors + per-code squared norms.
// ---------------------------------------------------------------------------
__global__ void vq_prep(const float* __restrict__ emb,
                        _Float16* __restrict__ embh,
                        _Float16* __restrict__ embt,
                        float* __restrict__ embsq) {
  int i = blockIdx.x * 256 + threadIdx.x;
  if (i < kNE * kDim) {
    float v = emb[i];
    embh[i] = (_Float16)v;
    int c = i >> 6, k = i & 63;
    embt[k * kNE + c] = (_Float16)v;
  }
  if (i < kNE) {
    float s = 0.0f;
    #pragma unroll 8
    for (int k = 0; k < kDim; ++k) { float v = emb[i * kDim + k]; s += v * v; }
    embsq[i] = s;
  }
}

// ---------------------------------------------------------------------------
// Main fused kernel: TDM z-tile load -> logits GEMM (WMMA) -> exp ->
//                    unnormalized mixing GEMM (WMMA) -> scale/argmax/loss/avgp
// ---------------------------------------------------------------------------
__global__ __launch_bounds__(256) void vq_main(
    const float* __restrict__ z,
    const _Float16* __restrict__ embh,
    const _Float16* __restrict__ embt,
    const float* __restrict__ embsq,
    float* __restrict__ out,
    float* __restrict__ ws_avgp,
    float* __restrict__ ws_loss) {
  extern __shared__ char smem[];
  const int tid  = threadIdx.x;
  const int wave = tid >> 5;
  const int lane = tid & 31;
  const int g    = lane >> 4;     // 16-lane half group
  const int lm   = lane & 15;
  const int row0 = blockIdx.x * kRowsPerWG + wave * kRowsPerWave;

  _Float16* ptile = (_Float16*)(smem + LDS_PT) + (size_t)wave * 16 * kNE;
  float*    ztile = (float*)(smem + LDS_ZT) + wave * (16 * kDim);
  float*    avgpl = (float*)(smem + LDS_AVGP);
  float*    rinvb = (float*)(smem + LDS_RINV);

  // ---- Tensor Data Mover: DMA this wave's 16x64 slice of z (C-strided in
  // global: element (k, m) at z[b*64*4096 + k*4096 + hw0 + m]) into LDS. ----
  {
    const int b   = row0 >> 12;
    const int hw0 = row0 & 4095;
    unsigned long long ga =
        (unsigned long long)(const void*)(z + ((size_t)b << 18) + hw0);
    unsigned ldsa = __builtin_amdgcn_groupstaticsize() + LDS_ZT +
                    (unsigned)wave * (16u * kDim * 4u);
    u32x4 d0;
    d0[0] = 1u;                                        // count = 1
    d0[1] = ldsa;                                      // lds_addr
    d0[2] = (unsigned)(ga & 0xffffffffull);            // global_addr[31:0]
    d0[3] = (unsigned)((ga >> 32) & 0x01ffffffull) | 0x80000000u; // [56:32]|type=2
    i32x8 d1;
    d1[0] = (int)(2u << 16);   // data_size = 4B
    d1[1] = (int)(16u << 16);  // tensor_dim0 = 16
    d1[2] = (int)(64u << 16);  // tensor_dim0 hi | tensor_dim1 = 64
    d1[3] = (int)(16u << 16);  // tensor_dim1 hi | tile_dim0 = 16
    d1[4] = 64;                // tile_dim1 = 64, tile_dim2 = 0
    d1[5] = 4096;              // tensor_dim0_stride = 4096 elements
    d1[6] = 0;
    d1[7] = 0;
    i32x4 dz4 = {0, 0, 0, 0};
    i32x8 dz8 = {0, 0, 0, 0, 0, 0, 0, 0};
    __builtin_amdgcn_tensor_load_to_lds(d0, d1, dz4, dz4, dz8, 0);
  }

  // zero the shared avg-prob accumulator while the DMA is in flight
  for (int c = tid; c < kNE; c += 256) avgpl[c] = 0.0f;
  __syncthreads();

  __builtin_amdgcn_s_wait_tensorcnt((short)0);
  asm volatile("" ::: "memory");

  // ---- build f16 A fragments of z (A layout: lane=row lm, K contiguous) ----
  v16h az0, az1;
  #pragma unroll
  for (int h = 0; h < 16; ++h) {
    int k0 = h + 16 * g;
    az0[h] = (_Float16)ztile[k0 * 16 + lm];
    az1[h] = (_Float16)ztile[(k0 + 32) * 16 + lm];
  }

  float rowsum[8], bmax[8];
  int   bidx[8];
  #pragma unroll
  for (int j = 0; j < 8; ++j) { rowsum[j] = 0.0f; bmax[j] = -3.4e38f; bidx[j] = 0; }
  v8f zq[4] = {};

  // ---- single fused sweep over the 1024 codes, 32 at a time ----
  #pragma unroll 1
  for (int cc = 0; cc < 32; ++cc) {
    const int c0 = cc * 32;
    #pragma unroll
    for (int t = 0; t < 2; ++t) {
      const int c = c0 + t * 16 + lm;              // this lane's code column
      // B layout: lane = column, K consecutive (16 per half-group)
      v16h b0 = *(const v16h*)(embh + c * kDim + 16 * g);
      v16h b1 = *(const v16h*)(embh + c * kDim + 32 + 16 * g);
      v8f acc = {};
      acc = wmma_f16(az0, b0, acc);
      acc = wmma_f16(az1, b1, acc);
      const float esq = embsq[c];
      #pragma unroll
      for (int j = 0; j < 8; ++j) {                // row m = j + 8g
        float lg = 2.0f * acc[j] - esq;            // softmax logit (row-const dropped)
        float p  = __expf(lg);
        rowsum[j] += p;
        if (lg > bmax[j]) { bmax[j] = lg; bidx[j] = c; }
        ptile[(j + 8 * g) * kNE + c] = (_Float16)p;
      }
    }
    asm volatile("s_wait_dscnt 0" ::: "memory");   // LDS round-trip = layout transpose
    v16h as;
    {
      const v8h lo = *(const v8h*)(ptile + lm * kNE + c0 + 16 * g);
      const v8h hi = *(const v8h*)(ptile + lm * kNE + c0 + 16 * g + 8);
      as = __builtin_shufflevector(lo, hi, 0, 1, 2, 3, 4, 5, 6, 7,
                                   8, 9, 10, 11, 12, 13, 14, 15);
    }
    #pragma unroll
    for (int tt = 0; tt < 4; ++tt) {               // 64 output dims = 4 N-tiles
      const int n = tt * 16 + lm;
      v16h bt = *(const v16h*)(embt + n * kNE + c0 + 16 * g);
      zq[tt] = wmma_f16(as, bt, zq[tt]);           // unnormalized z_q accumulate
    }
    asm volatile("" ::: "memory");
  }

  // ---- cross-lane reductions (16-lane groups): rowsum & argmax ----
  #pragma unroll
  for (int j = 0; j < 8; ++j) {
    #pragma unroll
    for (int off = 1; off < 16; off <<= 1) {
      rowsum[j] += __shfl_xor(rowsum[j], off, 32);
      float ov = __shfl_xor(bmax[j], off, 32);
      int   oi = __shfl_xor(bidx[j], off, 32);
      if (ov > bmax[j] || (ov == bmax[j] && oi < bidx[j])) {
        bmax[j] = ov; bidx[j] = oi;
      }
    }
  }
  float rinv[8];
  #pragma unroll
  for (int j = 0; j < 8; ++j) rinv[j] = 1.0f / rowsum[j];

  if (lm == 0) {
    #pragma unroll
    for (int j = 0; j < 8; ++j) {
      const int n = row0 + 8 * g + j;
      rinvb[wave * 16 + 8 * g + j] = rinv[j];
      out[OFF_IDX + (size_t)n] = (float)bidx[j];
      out[OFF_OH + (size_t)n * kNE + bidx[j]] = 1.0f;   // scatter into zeroed one-hot
    }
  }

  // ---- normalize, write z_q (B,C,H,W), accumulate loss ----
  float lsum = 0.0f;
  #pragma unroll
  for (int tt = 0; tt < 4; ++tt) {
    const int c  = tt * 16 + lm;
    const int n0 = row0 + 8 * g;
    const int b  = n0 >> 12;
    const int hw = n0 & 4095;
    const size_t base = ((size_t)(b * 64 + c) << 12) + hw;
    #pragma unroll
    for (int j = 0; j < 8; ++j) {                  // 8 consecutive hw positions
      float q  = zq[tt][j] * rinv[j];
      float zv = z[base + j];
      out[OFF_ZQ + base + j] = q;
      float d = q - zv;
      lsum += d * d;
    }
  }
  #pragma unroll
  for (int off = 1; off < 32; off <<= 1) lsum += __shfl_xor(lsum, off, 32);
  if (lane == 0) atomicAdd(ws_loss, lsum);

  // ---- normalized column sums from the cached p-tile -> avg_probs ----
  asm volatile("s_wait_dscnt 0" ::: "memory");
  float r16[16];
  #pragma unroll
  for (int m = 0; m < 16; ++m) r16[m] = rinvb[wave * 16 + m];
  const int e0 = lane * 32;                        // 32 columns per lane
  float colacc[32];
  #pragma unroll
  for (int i = 0; i < 32; ++i) colacc[i] = 0.0f;
  #pragma unroll 1
  for (int m = 0; m < 16; ++m) {
    const v8h* pr = (const v8h*)(ptile + m * kNE + e0);
    #pragma unroll
    for (int q = 0; q < 4; ++q) {
      v8h pv = pr[q];
      #pragma unroll
      for (int i = 0; i < 8; ++i) colacc[q * 8 + i] += (float)pv[i] * r16[m];
    }
  }
  #pragma unroll
  for (int i = 0; i < 32; ++i) atomicAdd(&avgpl[e0 + i], colacc[i]);

  __syncthreads();
  for (int c = tid; c < kNE; c += 256) atomicAdd(ws_avgp + c, avgpl[c]);
}

// ---------------------------------------------------------------------------
// Final: perplexity + loss scalars (single wave).
// ---------------------------------------------------------------------------
__global__ void vq_final(const float* __restrict__ ws_avgp,
                         const float* __restrict__ ws_loss,
                         float* __restrict__ out) {
  const int lane = threadIdx.x & 31;
  float h = 0.0f;
  for (int c = lane; c < kNE; c += 32) {
    float a = ws_avgp[c] * (1.0f / (float)kRows);
    h += a * __logf(a + 1e-10f);
  }
  #pragma unroll
  for (int off = 1; off < 32; off <<= 1) h += __shfl_xor(h, off, 32);
  if (lane == 0) {
    out[OFF_PERP] = __expf(-h);
    out[OFF_LOSS] = ws_loss[0] * (1.25f / (float)(kRows * kDim)); // (1+beta)*mean
  }
}

// ---------------------------------------------------------------------------
extern "C" void kernel_launch(void* const* d_in, const int* in_sizes, int n_in,
                              void* d_out, int out_size, void* d_ws, size_t ws_size,
                              hipStream_t stream) {
  (void)in_sizes; (void)n_in; (void)out_size; (void)ws_size;
  const float* z   = (const float*)d_in[0];
  const float* emb = (const float*)d_in[1];
  float* out = (float*)d_out;
  char*  ws  = (char*)d_ws;

  _Float16* embh  = (_Float16*)(ws + WS_EMBH);
  _Float16* embt  = (_Float16*)(ws + WS_EMBT);
  float*    embsq = (float*)(ws + WS_EMBSQ);
  float*    avgp  = (float*)(ws + WS_AVGP);
  float*    lossw = (float*)(ws + WS_LOSS);

  // deterministic per-call init: accumulators + the 537MB one-hot region
  (void)hipMemsetAsync(ws + WS_AVGP, 0, kNE * sizeof(float) + 16, stream);
  (void)hipMemsetAsync(out + OFF_OH, 0, (size_t)kRows * kNE * sizeof(float), stream);

  vq_prep<<<(kNE * kDim + 255) / 256, 256, 0, stream>>>(emb, embh, embt, embsq);
  vq_main<<<kNumWG, 256, LDS_TOTAL, stream>>>(z, embh, embt, embsq, out, avgp, lossw);
  vq_final<<<1, 32, 0, stream>>>(avgp, lossw, out);
}
